// MultiHeadDotProductAttention_19791209300575
// MI455X (gfx1250) — matrix operand order repaired and checked by reading, not compiled, not run on hardware
//
#include <hip/hip_runtime.h>
#include <hip/hip_bf16.h>

typedef __attribute__((ext_vector_type(16))) __bf16 v16bf;
typedef __attribute__((ext_vector_type(8)))  float  v8f;
typedef __attribute__((ext_vector_type(4)))  float  v4f;

// Problem constants (ViT-B/16 @384, EViT token pruning path)
#define BATCH 32
#define NTOK 577
#define CDIM 768
#define NH   12
#define HD   64
#define NPAD 592          // 37 * 16
#define NTILE 37
#define KEEP 404          // ceil(0.7 * 576)
#define ATT_SCALE 0.125f  // (768/12)^-0.5

// d_out element offsets (tuple flattened: x, index, ind, class_attention, final_tokens)
#define OFF_X     0
#define OFF_INDEX 14180352
#define OFF_IND   24109056
#define OFF_CLS   24121984
#define OFF_FT    24140416

// workspace byte offsets (total ~1.75 MB)
#define WS_USW   0u          // BATCH*24*32*16 bf16 = 786,432 B
#define WS_QB    786432u     // BATCH*NH float      =   1,536 B
#define WS_DOTS  787968u     // BATCH*NH*NPAD float =  909,312 B
#define WS_INDI  1697280u    // BATCH*KEEP int      =  51,712 B

// ---------------------------------------------------------------------------
// K1: stream x -> d_out copy (non-temporal stores: output is write-only for us,
// keeps x's lines resident in the 192MB L2 for k_dots' re-read). Also final_tokens.
__global__ void k_convert(const float* __restrict__ x, float* __restrict__ out) {
    int idx = blockIdx.x * blockDim.x + threadIdx.x;      // one float4 per thread
    v4f v = ((const v4f*)x)[idx];
    __builtin_nontemporal_store(v, ((v4f*)(out + OFF_X)) + idx);
    if (idx == 0) out[OFF_FT] = (float)KEEP;
}

// ---------------------------------------------------------------------------
// K2 (per batch): q_cls = x[b,0,:] @ W_q + b_q ; then U = Wk_head @ q_head,
// written directly in the WMMA B-operand register image (bf16), plus qb bias dots.
__global__ void k_prepU(const float* __restrict__ x, const float* __restrict__ Wqkv,
                        const float* __restrict__ bqkv,
                        __bf16* __restrict__ usw, float* __restrict__ qb) {
    __shared__ float xc[CDIM];
    __shared__ float q[CDIM];
    const int b = blockIdx.x, t = threadIdx.x;

    for (int i = t; i < CDIM; i += 256) xc[i] = x[(size_t)b * NTOK * CDIM + i];
    __syncthreads();

    for (int o = t; o < CDIM; o += 256) {
        float s = bqkv[o];
        for (int ci = 0; ci < CDIM; ++ci) s += xc[ci] * Wqkv[(size_t)ci * (3 * CDIM) + o];
        q[o] = s;
    }
    __syncthreads();

    // B-operand image: [kk][lane][e]; col=lane%16, h=lane/16: K=kk*32+8h+e (e<8), K=kk*32+16+8h+(e-8) (e>=8)
    for (int idx = t; idx < 24 * 32 * 16; idx += 256) {
        const int e    = idx & 15;
        const int lane = (idx >> 4) & 31;
        const int kk   = idx >> 9;
        const int head = lane & 15;
        __bf16 val = (__bf16)0.0f;
        if (head < NH) {
            const int ci = kk * 32 + ((lane < 16) ? 0 : 8) + ((e < 8) ? e : 16 + (e - 8));
            const float* wrow = Wqkv + (size_t)ci * (3 * CDIM) + CDIM + head * HD;
            const float* qh = q + head * HD;
            float s = 0.f;
            for (int dd = 0; dd < HD; ++dd) s += qh[dd] * wrow[dd];
            val = (__bf16)s;
        }
        usw[(size_t)b * (24 * 32 * 16) + idx] = val;
    }

    if (t < NH) {
        float s = 0.f;
        for (int dd = 0; dd < HD; ++dd) s += q[t * HD + dd] * bqkv[CDIM + t * HD + dd];
        qb[b * NH + t] = s;
    }
}

// ---------------------------------------------------------------------------
// K3: batched GEMM dots[b, j, h(16)] = X[b, j, :] @ U[b, :, h] via bf16 WMMA.
// A operand read as fp32 straight from x (L2-resident) and converted inline
// (v_cvt_pk_bf16_f32). Fully unrolled K-loop: 24 independent load groups give
// the scheduler a deep pipeline to hide L2 latency behind the WMMAs.
// One wave per 16-row tile. grid = (37 tiles, 32 batches), block = 32.
__global__ void k_dots(const float* __restrict__ x, const __bf16* __restrict__ usw,
                       const float* __restrict__ qb, float* __restrict__ dots) {
    const int tile = blockIdx.x, b = blockIdx.y;
    const int lane = threadIdx.x;
    int row = tile * 16 + (lane & 15);
    if (row > NTOK - 1) row = NTOK - 1;                  // clamp pad rows (ignored later)
    const int kbase = (lane < 16) ? 0 : 8;               // A-operand lane K pattern
    const float* arow = x + ((size_t)b * NTOK + row) * CDIM;

    v8f acc = {};
#pragma unroll
    for (int kk = 0; kk < 24; ++kk) {
        v4f f0 = *(const v4f*)(arow + kk * 32 + kbase);        // K = kbase+0..3
        v4f f1 = *(const v4f*)(arow + kk * 32 + kbase + 4);    // K = kbase+4..7
        v4f f2 = *(const v4f*)(arow + kk * 32 + kbase + 16);   // K = kbase+16..19
        v4f f3 = *(const v4f*)(arow + kk * 32 + kbase + 20);   // K = kbase+20..23
        v16bf a;
        a[0]  = (__bf16)f0[0]; a[1]  = (__bf16)f0[1]; a[2]  = (__bf16)f0[2]; a[3]  = (__bf16)f0[3];
        a[4]  = (__bf16)f1[0]; a[5]  = (__bf16)f1[1]; a[6]  = (__bf16)f1[2]; a[7]  = (__bf16)f1[3];
        a[8]  = (__bf16)f2[0]; a[9]  = (__bf16)f2[1]; a[10] = (__bf16)f2[2]; a[11] = (__bf16)f2[3];
        a[12] = (__bf16)f3[0]; a[13] = (__bf16)f3[1]; a[14] = (__bf16)f3[2]; a[15] = (__bf16)f3[3];
        v16bf bm = *(const v16bf*)(usw + (((size_t)b * 24 + kk) * 32 + lane) * 16);
        acc = __builtin_amdgcn_wmma_f32_16x16x32_bf16(
                  false, a, false, bm, (short)0, acc, false, false);
    }

    const int head = lane & 15;
    if (head < NH) {
        const float qbh = qb[b * NH + head];
        const int j0 = tile * 16 + ((lane >= 16) ? 8 : 0);     // D: VGPR r -> M=r / r+8
        float* dp = dots + ((size_t)b * NH + head) * NPAD + j0;
        v4f o0, o1;
        o0[0] = (acc[0] + qbh) * ATT_SCALE; o0[1] = (acc[1] + qbh) * ATT_SCALE;
        o0[2] = (acc[2] + qbh) * ATT_SCALE; o0[3] = (acc[3] + qbh) * ATT_SCALE;
        o1[0] = (acc[4] + qbh) * ATT_SCALE; o1[1] = (acc[5] + qbh) * ATT_SCALE;
        o1[2] = (acc[6] + qbh) * ATT_SCALE; o1[3] = (acc[7] + qbh) * ATT_SCALE;
        *(v4f*)(dp)     = o0;
        *(v4f*)(dp + 4) = o1;
    }
}

// ---------------------------------------------------------------------------
// K4: per-(b,h) softmax over j=0..576 (wave-level reductions), mean over heads.
__global__ void k_softmax(const float* __restrict__ dots, float* __restrict__ out) {
    __shared__ float att[NH][576];
    const int b = blockIdx.x;
    const int w = threadIdx.x >> 5;        // head = wave id (12 waves)
    const int lane = threadIdx.x & 31;
    const float* dp = dots + ((size_t)b * NH + w) * NPAD;

    float m = -1e30f;
    for (int j = lane; j < NTOK; j += 32) m = fmaxf(m, dp[j]);
    for (int o = 16; o > 0; o >>= 1) m = fmaxf(m, __shfl_xor(m, o, 32));

    float s = 0.f;
    for (int j = lane; j < NTOK; j += 32) s += __expf(dp[j] - m);
    for (int o = 16; o > 0; o >>= 1) s += __shfl_xor(s, o, 32);
    const float inv = 1.0f / s;

    for (int j = lane + 1; j < NTOK; j += 32)             // patch tokens only
        att[w][j - 1] = __expf(dp[j] - m) * inv;
    __syncthreads();

    for (int j = threadIdx.x; j < 576; j += 384) {
        float a = 0.f;
        for (int h = 0; h < NH; ++h) a += att[h][j];
        out[OFF_CLS + b * 576 + j] = a * (1.0f / NH);     // re-read by k_topk: keep RT
    }
}

// ---------------------------------------------------------------------------
// K5: exact top-k (k=404 of 576) with jax tie-breaking (value desc, index asc),
// output directly in ascending-index order via prefix count.
__global__ void k_topk(const float* __restrict__ cls_in, float* __restrict__ out,
                       int* __restrict__ indi) {
    __shared__ float v[576];
    __shared__ unsigned char sel[576];
    const int b = blockIdx.x, j = threadIdx.x;
    v[j] = cls_in[OFF_CLS + b * 576 + j];
    __syncthreads();
    const float vj = v[j];
    int rank = 0;
    for (int i = 0; i < 576; ++i) {
        const float vi = v[i];
        rank += (vi > vj) || (vi == vj && i < j);
    }
    sel[j] = (rank < KEEP) ? 1 : 0;
    __syncthreads();
    if (sel[j]) {
        int pos = 0;
        for (int i = 0; i < j; ++i) pos += sel[i];
        out[OFF_IND + b * KEEP + pos] = (float)j;
        indi[b * KEEP + pos] = j;
    }
}

// ---------------------------------------------------------------------------
// K6: broadcast ind over the channel dim: index[b, p, :] = ind[b, p]
// 192 threads x one v4f NT store each (write-only stream).
__global__ void k_bcast(const int* __restrict__ indi, float* __restrict__ out) {
    const int b = blockIdx.y, p = blockIdx.x;
    const float val = (float)indi[b * KEEP + p];
    v4f vv; vv[0] = val; vv[1] = val; vv[2] = val; vv[3] = val;
    v4f* dst = (v4f*)(out + OFF_INDEX + ((size_t)(b * KEEP + p)) * CDIM);
    __builtin_nontemporal_store(vv, dst + threadIdx.x);
}

// ---------------------------------------------------------------------------
extern "C" void kernel_launch(void* const* d_in, const int* in_sizes, int n_in,
                              void* d_out, int out_size, void* d_ws, size_t ws_size,
                              hipStream_t stream) {
    const float* x    = (const float*)d_in[0];
    // d_in[1] = keep_rate (0.7 -> pruning path, KEEP=404, compiled in)
    const float* Wqkv = (const float*)d_in[2];
    const float* bqkv = (const float*)d_in[3];
    // d_in[4], d_in[5] (W_out, b_out) are dead in the keep_rate<1 path.

    float* out = (float*)d_out;
    char* ws = (char*)d_ws;
    __bf16* usw  = (__bf16*)(ws + WS_USW);
    float*  qb   = (float*)(ws + WS_QB);
    float*  dots = (float*)(ws + WS_DOTS);
    int*    indi = (int*)(ws + WS_INDI);

    k_convert<<<dim3((BATCH * NTOK * CDIM) / 4 / 256), 256, 0, stream>>>(x, out);
    k_prepU  <<<dim3(BATCH),            256, 0, stream>>>(x, Wqkv, bqkv, usw, qb);
    k_dots   <<<dim3(NTILE, BATCH),      32, 0, stream>>>(x, usw, qb, dots);
    k_softmax<<<dim3(BATCH),            384, 0, stream>>>(dots, out);
    k_topk   <<<dim3(BATCH),            576, 0, stream>>>(out, out, indi);
    k_bcast  <<<dim3(KEEP, BATCH),      192, 0, stream>>>(indi, out);
}
